// TalkingHeadAttention_82145544503388
// MI455X (gfx1250) — matrix operand order repaired and checked by reading, not compiled, not run on hardware
//
#include <hip/hip_runtime.h>
#include <hip/hip_bf16.h>

typedef __attribute__((ext_vector_type(16))) _Float16 v16h;
typedef __attribute__((ext_vector_type(8)))  _Float16 v8h;
typedef __attribute__((ext_vector_type(8)))  float    v8f;
typedef unsigned int v4u __attribute__((ext_vector_type(4)));
typedef int          v4i __attribute__((ext_vector_type(4)));
typedef int          v8i __attribute__((ext_vector_type(8)));

#define BB   4
#define NN   1024
#define CC   768
#define HH   12
#define HD   64
#define SCL  0.125f      // 64^-0.5

#if __has_builtin(__builtin_amdgcn_tensor_load_to_lds)
#define USE_TDM 1
#else
#define USE_TDM 0
#endif

static __device__ inline v8f wmma16(v16h a, v16h b, v8f c) {
    return __builtin_amdgcn_wmma_f32_16x16x32_f16(
        false, a, false, b, (short)0, c, false, false);
}

// ---- fragment loaders (ISA 7.12.2 layouts, wave32) -------------------------
static __device__ inline v16h load_a_h(const _Float16* p0, int ld, int lane) {
    int m = lane & 15, hb = (lane >> 4) * 8;
    const _Float16* p = p0 + m * ld + hb;
    v8h lo = *(const v8h*)(p);
    v8h hi = *(const v8h*)(p + 16);
    v16h a;
#pragma unroll
    for (int i = 0; i < 8; ++i) { a[i] = lo[i]; a[i + 8] = hi[i]; }
    return a;
}
static __device__ inline v16h load_a_f32(const float* p0, int ld, int lane) {
    int m = lane & 15, hb = (lane >> 4) * 8;
    const float* p = p0 + m * ld + hb;
    v16h a;
#pragma unroll
    for (int i = 0; i < 8; ++i) { a[i] = (_Float16)p[i]; a[i + 8] = (_Float16)p[16 + i]; }
    return a;
}
// B (32x16): lane%16 = column n; elements = k in table order (LDS pitch 68)
static __device__ inline v16h load_b_f32(const float* p, int lane) {
    int hb = (lane >> 4) * 8;
    v16h b;
#pragma unroll
    for (int i = 0; i < 8; ++i) { b[i] = (_Float16)p[(hb + i) * 68]; b[i + 8] = (_Float16)p[(16 + hb + i) * 68]; }
    return b;
}

// ---- Tensor Data Mover: 2-D f32 tile (tile_w x tile_h) -> LDS --------------
// Pads 4 dwords after every 64 dwords stored -> 68-dword LDS row pitch.
#if USE_TDM
static __device__ inline void tdm_load_f32_tile(const float* gsrc, unsigned lds_off,
                                                int row_elems, int tile_w, int tile_h) {
    unsigned long long ga = (unsigned long long)(const void*)gsrc;
    v4u g0;
    g0[0] = 1u;                                        // count=1, user descriptor
    g0[1] = lds_off;                                   // lds_addr (bytes)
    g0[2] = (unsigned)(ga & 0xffffffffu);              // global_addr lo
    g0[3] = (unsigned)((ga >> 32) & 0x01ffffffu)       // global_addr hi
          | 0x80000000u;                               // type=2 ("image")
    v8i g1;
    g1[0] = (2 << 16)                                  // data_size = 4B
          | (1 << 20)                                  // pad_enable
          | (5 << 22)                                  // pad_interval = 64 dwords
          | (3 << 25);                                 // pad_amount  = 4 dwords
    g1[1] = (row_elems & 0xffff) << 16;                // tensor_dim0[15:0]
    g1[2] = (row_elems >> 16) & 0xffff;                // tensor_dim0[31:16]
    g1[3] = 16 | (tile_w << 16);                       // tensor_dim1=1<<20, tile_dim0
    g1[4] = tile_h;                                    // tile_dim1 (tile_dim2=0)
    g1[5] = row_elems;                                 // tensor_dim0_stride[31:0]
    g1[6] = 0;
    g1[7] = 0;
    v4i z4 = {};
    v8i z8 = {};
    __builtin_amdgcn_tensor_load_to_lds(g0, g1, z4, z4, z8, 0);
}
#endif

// ============================================================================
// Kernel 1: qkv = X @ Wqkv ; scatter into Q(scaled), K^T, V (all f16)
// 8 waves/block share a 64-wide n-block; W tile staged in LDS by TDM.
// ============================================================================
__global__ __launch_bounds__(256)
void qkv_kernel(const float* __restrict__ X, const float* __restrict__ Wq,
                _Float16* __restrict__ Qs, _Float16* __restrict__ Kt,
                _Float16* __restrict__ V) {
    __shared__ float wt[2][32][68];                    // padded pitch (bank spread)
    int lane = threadIdx.x & 31, w = threadIdx.x >> 5;
    int row0 = (blockIdx.y * 8 + w) * 16;
    int col0 = blockIdx.x * 64;
    v8f acc[4] = {};

#if USE_TDM
    if (w == 0)
        tdm_load_f32_tile(Wq + col0, (unsigned)(unsigned long long)(uintptr_t)&wt[0][0][0],
                          3 * CC, 64, 32);
#endif
    for (int s = 0; s < CC / 32; ++s) {
        int kc = s * 32;
#if USE_TDM
        if (w == 0) {
            if (s + 1 < CC / 32) {
                tdm_load_f32_tile(Wq + (size_t)(kc + 32) * (3 * CC) + col0,
                                  (unsigned)(unsigned long long)(uintptr_t)&wt[(s + 1) & 1][0][0],
                                  3 * CC, 64, 32);
                __builtin_amdgcn_s_wait_tensorcnt(1);   // current tile complete
            } else {
                __builtin_amdgcn_s_wait_tensorcnt(0);
            }
        }
        __syncthreads();
        const float* wrow = &wt[s & 1][0][0];
#else
        __syncthreads();
        {   // cooperative staging fallback: 8 contiguous f32 per thread
            int t = threadIdx.x, r = t >> 3, c = (t & 7) * 8;
#pragma unroll
            for (int i = 0; i < 8; ++i)
                wt[0][r][c + i] = Wq[(size_t)(kc + r) * (3 * CC) + col0 + c + i];
        }
        __syncthreads();
        const float* wrow = &wt[0][0][0];
#endif
        v16h a = load_a_f32(X + (size_t)row0 * CC + kc, CC, lane);
#pragma unroll
        for (int j = 0; j < 4; ++j) {
            v16h b = load_b_f32(wrow + (lane & 15) + 16 * j, lane);
            acc[j] = wmma16(a, b, acc[j]);
        }
        __syncthreads();
    }

    int part = col0 / CC;
    int h    = (col0 % CC) / HD;
    int bb   = row0 / NN, tok0 = row0 % NN;
    int m0 = (lane >> 4) * 8, n = lane & 15;
    size_t hbase = (size_t)(bb * HH + h);
    if (part == 1) {               // K: store transposed, 8 contiguous tokens/lane
#pragma unroll
        for (int j = 0; j < 4; ++j) {
            v8h v;
#pragma unroll
            for (int r = 0; r < 8; ++r) v[r] = (_Float16)acc[j][r];
            *(v8h*)(Kt + (hbase * HD + 16 * j + n) * NN + tok0 + m0) = v;
        }
    } else {
        _Float16* dst = (part == 0) ? Qs : V;
        float s = (part == 0) ? SCL : 1.0f;
#pragma unroll
        for (int j = 0; j < 4; ++j)
#pragma unroll
            for (int r = 0; r < 8; ++r)
                dst[(hbase * NN + tok0 + m0 + r) * HD + 16 * j + n] =
                    (_Float16)(acc[j][r] * s);
    }
}

// ============================================================================
// Kernel 2: fused talking-heads attention core.
// 1 block = (batch, 16-query tile); 12 waves = 12 heads; two-pass softmax.
// ============================================================================
__global__ __launch_bounds__(384)
void attn_kernel(const _Float16* __restrict__ Qs, const _Float16* __restrict__ Kt,
                 const _Float16* __restrict__ V,
                 const float* __restrict__ w_l, const float* __restrict__ b_l,
                 const float* __restrict__ w_w, const float* __restrict__ b_w,
                 _Float16* __restrict__ Xo) {
    __shared__ float    sS[HH][2][8][32];   // frag-space score/P exchange (24KB)
    __shared__ _Float16 sP[HH][16][32];     // per-wave P transpose staging (12KB)

    int lane = threadIdx.x & 31, g = threadIdx.x >> 5;
    int bb = blockIdx.x >> 6;
    int q0 = (blockIdx.x & 63) * 16;
    size_t hb = (size_t)(bb * HH + g);

    float wl[HH], ww[HH];
#pragma unroll
    for (int h = 0; h < HH; ++h) { wl[h] = w_l[h * HH + g]; ww[h] = w_w[h * HH + g]; }
    float bl = b_l[g], bw = b_w[g];

    v16h qa0 = load_a_h(Qs + (hb * NN + q0) * HD,      HD, lane);
    v16h qa1 = load_a_h(Qs + (hb * NN + q0) * HD + 32, HD, lane);

    float rowM[8], rowZ[8];
#pragma unroll
    for (int r = 0; r < 8; ++r) { rowM[r] = -1e30f; rowZ[r] = 0.f; }

    // ---------------- pass 1: mixed scores -> online max/sum ----------------
    for (int ks = 0; ks < NN; ks += 32) {
        if (ks + 32 < NN) {   // stream-prefetch next K rows
            __builtin_prefetch(Kt + (hb * HD +  0 + lane) * NN + ks + 32, 0, 1);
            __builtin_prefetch(Kt + (hb * HD + 32 + lane) * NN + ks + 32, 0, 1);
        }
#pragma unroll
        for (int sub = 0; sub < 2; ++sub) {
            int kt0 = ks + 16 * sub;
            v16h b0, b1;
#pragma unroll
            for (int i = 0; i < 16; ++i) {
                int kd = ((i & 8) << 1) + ((lane >> 4) << 3) + (i & 7);
                b0[i] = Kt[(hb * HD +  0 + kd) * NN + kt0 + (lane & 15)];
                b1[i] = Kt[(hb * HD + 32 + kd) * NN + kt0 + (lane & 15)];
            }
            v8f s = {};
            s = wmma16(qa0, b0, s);
            s = wmma16(qa1, b1, s);
#pragma unroll
            for (int r = 0; r < 8; ++r) sS[g][sub][r][lane] = s[r];
        }
        __syncthreads();
        float t0[8], t1[8];
#pragma unroll
        for (int r = 0; r < 8; ++r) {
            float a = bl, b = bl;
#pragma unroll
            for (int h = 0; h < HH; ++h) {
                a += sS[h][0][r][lane] * wl[h];
                b += sS[h][1][r][lane] * wl[h];
            }
            t0[r] = a; t1[r] = b;
        }
#pragma unroll
        for (int r = 0; r < 8; ++r) {
            float tm = fmaxf(t0[r], t1[r]);
            for (int d = 1; d < 16; d <<= 1) tm = fmaxf(tm, __shfl_xor(tm, d, 16));
            float mn = fmaxf(rowM[r], tm);
            float ps = __expf(t0[r] - mn) + __expf(t1[r] - mn);
            for (int d = 1; d < 16; d <<= 1) ps += __shfl_xor(ps, d, 16);
            rowZ[r] = rowZ[r] * __expf(rowM[r] - mn) + ps;
            rowM[r] = mn;
        }
        __syncthreads();
    }

    float invZ[8];
#pragma unroll
    for (int r = 0; r < 8; ++r) invZ[r] = 1.0f / rowZ[r];

    // ---------------- pass 2: P, second mix, @V -----------------------------
    v8f o[4] = {};
    for (int ks = 0; ks < NN; ks += 32) {
        if (ks + 32 < NN) {
            __builtin_prefetch(Kt + (hb * HD +  0 + lane) * NN + ks + 32, 0, 1);
            __builtin_prefetch(V + (hb * NN + ks + 32 + lane) * HD, 0, 1);
        }
#pragma unroll
        for (int sub = 0; sub < 2; ++sub) {
            int kt0 = ks + 16 * sub;
            v16h b0, b1;
#pragma unroll
            for (int i = 0; i < 16; ++i) {
                int kd = ((i & 8) << 1) + ((lane >> 4) << 3) + (i & 7);
                b0[i] = Kt[(hb * HD +  0 + kd) * NN + kt0 + (lane & 15)];
                b1[i] = Kt[(hb * HD + 32 + kd) * NN + kt0 + (lane & 15)];
            }
            v8f s = {};
            s = wmma16(qa0, b0, s);
            s = wmma16(qa1, b1, s);
#pragma unroll
            for (int r = 0; r < 8; ++r) sS[g][sub][r][lane] = s[r];
        }
        __syncthreads();
        float t0[8], t1[8];
#pragma unroll
        for (int r = 0; r < 8; ++r) {
            float a = bl, b = bl;
#pragma unroll
            for (int h = 0; h < HH; ++h) {
                a += sS[h][0][r][lane] * wl[h];
                b += sS[h][1][r][lane] * wl[h];
            }
            t0[r] = a; t1[r] = b;
        }
        __syncthreads();
#pragma unroll
        for (int r = 0; r < 8; ++r) {
            sS[g][0][r][lane] = __expf(t0[r] - rowM[r]) * invZ[r];
            sS[g][1][r][lane] = __expf(t1[r] - rowM[r]) * invZ[r];
        }
        __syncthreads();
        int m0 = (lane >> 4) * 8, n = lane & 15;
#pragma unroll
        for (int r = 0; r < 8; ++r) {
            float a = bw, b = bw;
#pragma unroll
            for (int h = 0; h < HH; ++h) {
                a += sS[h][0][r][lane] * ww[h];
                b += sS[h][1][r][lane] * ww[h];
            }
            sP[g][m0 + r][n]      = (_Float16)a;
            sP[g][m0 + r][16 + n] = (_Float16)b;
        }
        v16h pa = load_a_h(&sP[g][0][0], 32, lane);   // same-wave LDS RAW: hw-waited
#pragma unroll
        for (int j = 0; j < 4; ++j) {
            v16h vb;
#pragma unroll
            for (int i = 0; i < 16; ++i)
                vb[i] = V[(hb * NN + ks + ((i & 8) << 1) + ((lane >> 4) << 3) + (i & 7)) * HD + 16 * j + (lane & 15)];
            o[j] = wmma16(pa, vb, o[j]);
        }
        __syncthreads();
    }

    int m0 = (lane >> 4) * 8, n = lane & 15;
#pragma unroll
    for (int j = 0; j < 4; ++j)
#pragma unroll
        for (int r = 0; r < 8; ++r)
            Xo[((size_t)bb * NN + q0 + m0 + r) * CC + g * HD + 16 * j + n] =
                (_Float16)o[j][r];
}

// ============================================================================
// Kernel 3: out = Xo @ Wproj + b_proj  (f16 x f16 -> f32), TDM-staged weights
// ============================================================================
__global__ __launch_bounds__(256)
void proj_kernel(const _Float16* __restrict__ Xo, const float* __restrict__ Wp,
                 const float* __restrict__ bp, float* __restrict__ out) {
    __shared__ float wt[2][32][68];
    int lane = threadIdx.x & 31, w = threadIdx.x >> 5;
    int row0 = (blockIdx.y * 8 + w) * 16;
    int col0 = blockIdx.x * 64;
    v8f acc[4] = {};

#if USE_TDM
    if (w == 0)
        tdm_load_f32_tile(Wp + col0, (unsigned)(unsigned long long)(uintptr_t)&wt[0][0][0],
                          CC, 64, 32);
#endif
    for (int s = 0; s < CC / 32; ++s) {
        int kc = s * 32;
#if USE_TDM
        if (w == 0) {
            if (s + 1 < CC / 32) {
                tdm_load_f32_tile(Wp + (size_t)(kc + 32) * CC + col0,
                                  (unsigned)(unsigned long long)(uintptr_t)&wt[(s + 1) & 1][0][0],
                                  CC, 64, 32);
                __builtin_amdgcn_s_wait_tensorcnt(1);
            } else {
                __builtin_amdgcn_s_wait_tensorcnt(0);
            }
        }
        __syncthreads();
        const float* wrow = &wt[s & 1][0][0];
#else
        __syncthreads();
        {
            int t = threadIdx.x, r = t >> 3, c = (t & 7) * 8;
#pragma unroll
            for (int i = 0; i < 8; ++i)
                wt[0][r][c + i] = Wp[(size_t)(kc + r) * CC + col0 + c + i];
        }
        __syncthreads();
        const float* wrow = &wt[0][0][0];
#endif
        v16h a = load_a_h(Xo + (size_t)row0 * CC + kc, CC, lane);
#pragma unroll
        for (int j = 0; j < 4; ++j) {
            v16h b = load_b_f32(wrow + (lane & 15) + 16 * j, lane);
            acc[j] = wmma16(a, b, acc[j]);
        }
        __syncthreads();
    }
    int m0 = (lane >> 4) * 8, n = lane & 15;
#pragma unroll
    for (int j = 0; j < 4; ++j)
#pragma unroll
        for (int r = 0; r < 8; ++r)
            out[(size_t)(row0 + m0 + r) * CC + col0 + 16 * j + n] =
                acc[j][r] + bp[col0 + 16 * j + n];
}

// ============================================================================
extern "C" void kernel_launch(void* const* d_in, const int* in_sizes, int n_in,
                              void* d_out, int out_size, void* d_ws, size_t ws_size,
                              hipStream_t stream) {
    const float* X   = (const float*)d_in[0];
    const float* Wq  = (const float*)d_in[1];
    const float* w_l = (const float*)d_in[2];
    const float* b_l = (const float*)d_in[3];
    const float* w_w = (const float*)d_in[4];
    const float* b_w = (const float*)d_in[5];
    const float* Wp  = (const float*)d_in[6];
    const float* bp  = (const float*)d_in[7];
    float* out = (float*)d_out;

    const size_t headsz = (size_t)BB * HH * NN * HD * sizeof(_Float16); // 6 MB
    char* ws = (char*)d_ws;
    _Float16* Qs = (_Float16*)(ws);
    _Float16* Kt = (_Float16*)(ws + headsz);
    _Float16* V  = (_Float16*)(ws + 2 * headsz);
    _Float16* Xo = (_Float16*)(ws + 3 * headsz);

    qkv_kernel<<<dim3((3 * CC) / 64, (BB * NN) / (16 * 8)), 256, 0, stream>>>(
        X, Wq, Qs, Kt, V);
    attn_kernel<<<BB * (NN / 16), HH * 32, 0, stream>>>(
        Qs, Kt, V, w_l, b_l, w_w, b_w, Xo);
    proj_kernel<<<dim3(CC / 64, (BB * NN) / (16 * 8)), 256, 0, stream>>>(
        Xo, Wp, bp, out);
}
